// SelfAttention_85890755986056
// MI455X (gfx1250) — matrix operand order repaired and checked by reading, not compiled, not run on hardware
//
#include <hip/hip_runtime.h>
#include <hip/hip_bf16.h>
#include <stdint.h>

typedef __bf16 v8bf  __attribute__((ext_vector_type(8)));
typedef __bf16 v16bf __attribute__((ext_vector_type(16)));
typedef float  v8f   __attribute__((ext_vector_type(8)));

#define D_MODEL 1024
#define NUM_HEADS 16
#define BATCH 4
#define SEQ 2048
#define HD 64                       // head dim
#define MROWS (BATCH * SEQ)         // 8192 token rows
#define HEADS (BATCH * NUM_HEADS)   // 64 flat heads
#define HSIZE (SEQ * HD)            // 131072 elems per head slab (= 2^17)

// ---------------------------------------------------------------------------
// WMMA fragment loaders (bf16, 16x16x32 shape).
// A 16x32 (MxK): lane l -> row m=l&15, g=l>>4; elems[0..7]=K(g*8..), [8..15]=K(16+g*8..)
// B 32x16 (KxN): lane l -> col n=l&15, g=l>>4; elems[0..7]=K(g*8..), [8..15]=K(16+g*8..)
// ---------------------------------------------------------------------------
__device__ __forceinline__ v16bf load_fragA(const __bf16* base, int stride, int lane) {
    const int m = lane & 15, g = lane >> 4;
    const __bf16* p = base + (size_t)m * stride + g * 8;
    v8bf lo = *reinterpret_cast<const v8bf*>(p);
    v8bf hi = *reinterpret_cast<const v8bf*>(p + 16);
    return __builtin_shufflevector(lo, hi, 0, 1, 2, 3, 4, 5, 6, 7,
                                   8, 9, 10, 11, 12, 13, 14, 15);
}

__device__ __forceinline__ v16bf load_fragB(const __bf16* base, int stride, int lane) {
    const int n = lane & 15, g = lane >> 4;
    const __bf16* p = base + (size_t)n * stride + g * 8;
    v8bf lo = *reinterpret_cast<const v8bf*>(p);
    v8bf hi = *reinterpret_cast<const v8bf*>(p + 16);
    return __builtin_shufflevector(lo, hi, 0, 1, 2, 3, 4, 5, 6, 7,
                                   8, 9, 10, 11, 12, 13, 14, 15);
}

// ---------------------------------------------------------------------------
// CDNA5 async global->LDS copy (16B per active lane), tracked by ASYNCcnt.
// ---------------------------------------------------------------------------
__device__ __forceinline__ void async_b128(uint32_t lds_off, const void* g) {
    asm volatile("global_load_async_to_lds_b128 %0, %1, off"
                 :: "v"(lds_off), "v"((unsigned long long)(uintptr_t)g)
                 : "memory");
}
#define WAIT_ASYNC(n) asm volatile("s_wait_asynccnt " #n ::: "memory")

// ---------------------------------------------------------------------------
// fp32 -> bf16 conversion (grid-stride)
// ---------------------------------------------------------------------------
__global__ void f32_to_bf16(const float* __restrict__ in,
                            __bf16* __restrict__ out, int n) {
    int i = blockIdx.x * blockDim.x + threadIdx.x;
    const int stride = gridDim.x * blockDim.x;
    for (; i < n; i += stride) out[i] = (__bf16)in[i];
}

// ---------------------------------------------------------------------------
// QKV projection: y[m,n] = sum_k x[m,k] * W[n,k] + bias[n]
// One wave per 32x64 output tile (2 A-frags reuse 4 B-frags -> 8 WMMA / chunk).
// vtrans==1 writes V transposed per head slab: Vt[bh, d, s] (s contiguous).
// ---------------------------------------------------------------------------
__global__ void __launch_bounds__(32)
qkv_proj(const __bf16* __restrict__ xb, const __bf16* __restrict__ Wb,
         const float* __restrict__ bias, __bf16* __restrict__ out, int vtrans) {
    const int lane = threadIdx.x & 31;
    const int m0 = blockIdx.x * 32;
    const int n0 = blockIdx.y * 64;

    v8f acc0[4] = {};
    v8f acc1[4] = {};
    for (int kc = 0; kc < D_MODEL; kc += 32) {
        const v16bf a0 = load_fragA(xb + (size_t)m0 * D_MODEL + kc, D_MODEL, lane);
        const v16bf a1 = load_fragA(xb + (size_t)(m0 + 16) * D_MODEL + kc, D_MODEL, lane);
#pragma unroll
        for (int nt = 0; nt < 4; ++nt) {
            const v16bf b =
                load_fragB(Wb + (size_t)(n0 + nt * 16) * D_MODEL + kc, D_MODEL, lane);
            acc0[nt] = __builtin_amdgcn_wmma_f32_16x16x32_bf16(
                false, a0, false, b, (short)0, acc0[nt], false, false);
            acc1[nt] = __builtin_amdgcn_wmma_f32_16x16x32_bf16(
                false, a1, false, b, (short)0, acc1[nt], false, false);
        }
    }

    const int g = lane >> 4;
    const int nl = lane & 15;
#pragma unroll
    for (int mt = 0; mt < 2; ++mt) {
        const v8f* acc = mt ? acc1 : acc0;
#pragma unroll
        for (int nt = 0; nt < 4; ++nt) {
            const int n = n0 + nt * 16 + nl;
            const float bv = bias[n];
#pragma unroll
            for (int v = 0; v < 8; ++v) {
                const int r = m0 + mt * 16 + v + g * 8;
                const __bf16 hb = (__bf16)(acc[nt][v] + bv);
                const size_t flat = (size_t)r * D_MODEL + n;
                if (vtrans) {
                    const size_t bh = flat >> 17;            // / HSIZE
                    const size_t rem = flat & (HSIZE - 1);
                    const size_t s2 = rem >> 6;              // / HD
                    const size_t d2 = rem & (HD - 1);
                    out[bh * HSIZE + d2 * SEQ + s2] = hb;
                } else {
                    out[flat] = hb;
                }
            }
        }
    }
}

// ---------------------------------------------------------------------------
// Flash attention with async-LDS K/V staging.
// Block = 4 waves = 64 query rows of one head. K/V streamed in 32-key chunks,
// double-buffered via GLOBAL_LOAD_ASYNC_TO_LDS_B128 (ASYNCcnt), shared by all
// 4 waves. Each wave: 16 q rows, online softmax, P transposed via LDS.
// ---------------------------------------------------------------------------
__global__ void __launch_bounds__(128)
flash_attn(const __bf16* __restrict__ Qb, const __bf16* __restrict__ Kb,
           const __bf16* __restrict__ Vt, float* __restrict__ out) {
    __shared__ __bf16 kbuf[2][32 * HD];   // 32 keys x 64 dims (row stride 64)
    __shared__ __bf16 vbuf[2][HD * 32];   // 64 dims x 32 keys (row stride 32)
    __shared__ __bf16 plds[4][16 * 32];   // per-wave P transpose scratch

    const int tid = threadIdx.x;
    const int lane = tid & 31;
    const int wv = tid >> 5;
    const int head = blockIdx.x >> 5;          // 32 blocks per head
    const int q0 = (blockIdx.x & 31) * 64 + wv * 16;

    const __bf16* qh = Qb + (size_t)head * HSIZE + (size_t)q0 * HD;
    const __bf16* kh = Kb + (size_t)head * HSIZE;
    const __bf16* vh = Vt + (size_t)head * HSIZE;

    const v16bf aq0 = load_fragA(qh, HD, lane);       // head dims 0..31
    const v16bf aq1 = load_fragA(qh + 32, HD, lane);  // head dims 32..63

    v8f acc[4] = {};
    float mrow[8], lrow[8];
#pragma unroll
    for (int v = 0; v < 8; ++v) { mrow[v] = -1e30f; lrow[v] = 0.0f; }

    const float sc = 0.03125f;  // 1/sqrt(D_MODEL) per reference
    const int g = lane >> 4;
    const int nl = lane & 15;
    __bf16* pw = &plds[wv][0];

    // ---- stage chunk 0 --------------------------------------------------
    // K chunk: 4 KB fully contiguous = 256 x 16B units; V chunk: 64 rows x 64B.
    {
        const char* kg = (const char*)kh;
        const char* vg = (const char*)vh;
        const uint32_t kl = (uint32_t)(uintptr_t)&kbuf[0][0];
        const uint32_t vl = (uint32_t)(uintptr_t)&vbuf[0][0];
#pragma unroll
        for (int u = 0; u < 2; ++u) {
            const int unit = tid + u * 128;
            async_b128(kl + unit * 16, kg + unit * 16);
        }
#pragma unroll
        for (int u = 0; u < 2; ++u) {
            const int unit = tid + u * 128;
            async_b128(vl + unit * 16,
                       vg + (size_t)(unit >> 2) * (SEQ * 2) + (unit & 3) * 16);
        }
    }

    for (int kc = 0; kc < SEQ; kc += 32) {
        const int cur = (kc >> 5) & 1;
        if (kc + 32 < SEQ) {
            // stage next chunk into the other buffer (it was consumed last iteration)
            const char* kg = (const char*)(kh + (size_t)(kc + 32) * HD);
            const char* vg = (const char*)(vh + (kc + 32));
            const uint32_t kl = (uint32_t)(uintptr_t)&kbuf[cur ^ 1][0];
            const uint32_t vl = (uint32_t)(uintptr_t)&vbuf[cur ^ 1][0];
#pragma unroll
            for (int u = 0; u < 2; ++u) {
                const int unit = tid + u * 128;
                async_b128(kl + unit * 16, kg + unit * 16);
            }
#pragma unroll
            for (int u = 0; u < 2; ++u) {
                const int unit = tid + u * 128;
                async_b128(vl + unit * 16,
                           vg + (size_t)(unit >> 2) * (SEQ * 2) + (unit & 3) * 16);
            }
            WAIT_ASYNC(0x4);   // in-order: current chunk's 4 ops have landed
        } else {
            WAIT_ASYNC(0x0);
        }
        __syncthreads();       // current buffer visible to all 4 waves

        const __bf16* kb = &kbuf[cur][0];
        const __bf16* vb = &vbuf[cur][0];

        // scores: S[16x32] = Q[16x64] x K[64x32]^T  (4 WMMAs, B frags from LDS)
        v8f s0 = {}, s1 = {};
        v16bf bk = load_fragB(kb, HD, lane);
        s0 = __builtin_amdgcn_wmma_f32_16x16x32_bf16(false, aq0, false, bk,
                                                     (short)0, s0, false, false);
        bk = load_fragB(kb + 32, HD, lane);
        s0 = __builtin_amdgcn_wmma_f32_16x16x32_bf16(false, aq1, false, bk,
                                                     (short)0, s0, false, false);
        bk = load_fragB(kb + 16 * HD, HD, lane);
        s1 = __builtin_amdgcn_wmma_f32_16x16x32_bf16(false, aq0, false, bk,
                                                     (short)0, s1, false, false);
        bk = load_fragB(kb + 16 * HD + 32, HD, lane);
        s1 = __builtin_amdgcn_wmma_f32_16x16x32_bf16(false, aq1, false, bk,
                                                     (short)0, s1, false, false);

        // online softmax across this 32-key chunk (row = v + 8*g, col = nl)
        float mnew[8], p0[8], p1[8];
#pragma unroll
        for (int v = 0; v < 8; ++v) {
            s0[v] *= sc;
            s1[v] *= sc;
            float mx = fmaxf(s0[v], s1[v]);
            mx = fmaxf(mx, __shfl_xor(mx, 1, 16));
            mx = fmaxf(mx, __shfl_xor(mx, 2, 16));
            mx = fmaxf(mx, __shfl_xor(mx, 4, 16));
            mx = fmaxf(mx, __shfl_xor(mx, 8, 16));
            mnew[v] = fmaxf(mrow[v], mx);
        }
#pragma unroll
        for (int v = 0; v < 8; ++v) {
            const float corr = __expf(mrow[v] - mnew[v]);
            p0[v] = __expf(s0[v] - mnew[v]);
            p1[v] = __expf(s1[v] - mnew[v]);
            float rs = p0[v] + p1[v];
            rs += __shfl_xor(rs, 1, 16);
            rs += __shfl_xor(rs, 2, 16);
            rs += __shfl_xor(rs, 4, 16);
            rs += __shfl_xor(rs, 8, 16);
            lrow[v] = lrow[v] * corr + rs;
            mrow[v] = mnew[v];
#pragma unroll
            for (int dt = 0; dt < 4; ++dt) acc[dt][v] *= corr;
        }

        // transpose P (C-layout -> A-layout) through LDS, intra-wave only
#pragma unroll
        for (int v = 0; v < 8; ++v) {
            const int r = v + g * 8;
            pw[r * 32 + nl] = (__bf16)p0[v];
            pw[r * 32 + nl + 16] = (__bf16)p1[v];
        }
        const __bf16* pl = pw + (lane & 15) * 32 + g * 8;
        const v8bf lo = *reinterpret_cast<const v8bf*>(pl);
        const v8bf hi = *reinterpret_cast<const v8bf*>(pl + 16);
        const v16bf pa = __builtin_shufflevector(lo, hi, 0, 1, 2, 3, 4, 5, 6, 7,
                                                 8, 9, 10, 11, 12, 13, 14, 15);

        // O[16x64] += P[16x32] x V[32x64]  (4 WMMAs, Vt chunk in LDS, stride 32)
#pragma unroll
        for (int dt = 0; dt < 4; ++dt) {
            const v16bf bv = load_fragB(vb + (size_t)(dt * 16) * 32, 32, lane);
            acc[dt] = __builtin_amdgcn_wmma_f32_16x16x32_bf16(
                false, pa, false, bv, (short)0, acc[dt], false, false);
        }
        __syncthreads();       // all waves done reading before buffer is re-staged
    }

    // normalize by row sums and store fp32 (flat view == output layout)
    float* oh = out + (size_t)head * HSIZE + (size_t)q0 * HD;
#pragma unroll
    for (int dt = 0; dt < 4; ++dt) {
#pragma unroll
        for (int v = 0; v < 8; ++v) {
            const int r = v + g * 8;
            oh[(size_t)r * HD + dt * 16 + nl] = acc[dt][v] / lrow[v];
        }
    }
}

// ---------------------------------------------------------------------------
// Host-side launcher
// ---------------------------------------------------------------------------
extern "C" void kernel_launch(void* const* d_in, const int* in_sizes, int n_in,
                              void* d_out, int out_size, void* d_ws, size_t ws_size,
                              hipStream_t stream) {
    const float* x  = (const float*)d_in[0];
    const float* Wq = (const float*)d_in[1];
    const float* bq = (const float*)d_in[2];
    const float* Wk = (const float*)d_in[3];
    const float* bk = (const float*)d_in[4];
    const float* Wv = (const float*)d_in[5];
    const float* bv = (const float*)d_in[6];
    float* out = (float*)d_out;

    // workspace carve-up (bf16): x, Wq, Wk, Wv, Q, K, Vt  (~70 MB total)
    __bf16* p = (__bf16*)d_ws;
    __bf16* xb  = p; p += (size_t)MROWS * D_MODEL;
    __bf16* Wqb = p; p += (size_t)D_MODEL * D_MODEL;
    __bf16* Wkb = p; p += (size_t)D_MODEL * D_MODEL;
    __bf16* Wvb = p; p += (size_t)D_MODEL * D_MODEL;
    __bf16* Qb  = p; p += (size_t)MROWS * D_MODEL;
    __bf16* Kb  = p; p += (size_t)MROWS * D_MODEL;
    __bf16* Vt  = p; p += (size_t)MROWS * D_MODEL;

    f32_to_bf16<<<4096, 256, 0, stream>>>(x, xb, MROWS * D_MODEL);
    f32_to_bf16<<<1024, 256, 0, stream>>>(Wq, Wqb, D_MODEL * D_MODEL);
    f32_to_bf16<<<1024, 256, 0, stream>>>(Wk, Wkb, D_MODEL * D_MODEL);
    f32_to_bf16<<<1024, 256, 0, stream>>>(Wv, Wvb, D_MODEL * D_MODEL);

    dim3 gproj(MROWS / 32, D_MODEL / 64);   // 256 x 16 tiles of 32x64
    qkv_proj<<<gproj, 32, 0, stream>>>(xb, Wqb, bq, Qb, 0);
    qkv_proj<<<gproj, 32, 0, stream>>>(xb, Wkb, bk, Kb, 0);
    qkv_proj<<<gproj, 32, 0, stream>>>(xb, Wvb, bv, Vt, 1);

    // 64 heads * 32 blocks, each block = 4 waves = 64 query rows
    flash_attn<<<HEADS * (SEQ / 64), 128, 0, stream>>>(Qb, Kb, Vt, out);
}
